// RoboticsDiTLayer_39926015984295
// MI455X (gfx1250) — matrix operand order repaired and checked by reading, not compiled, not run on hardware
//
#include <hip/hip_runtime.h>

// ---------------------------------------------------------------------------
// Problem constants (from reference): B=4, S=4096, D=2048
// ---------------------------------------------------------------------------
constexpr int BB = 4;
constexpr int SS = 4096;
constexpr int DD = 2048;
constexpr int MM = BB * SS;       // 16384 rows
constexpr int KK = DD;            // 2048 reduction dim
constexpr int N3 = 3 * DD;        // 6144 controller outputs
constexpr float EPSV = 1e-6f;
constexpr float DTV  = 0.1f;

typedef __attribute__((ext_vector_type(16))) __bf16 v16bf;
typedef __attribute__((ext_vector_type(8)))  __bf16 bf16x8;
typedef __attribute__((ext_vector_type(8)))  float  v8f;
typedef int v4i __attribute__((vector_size(16)));   // matches builtin param type

#define AS1 __attribute__((address_space(1)))
#define AS3 __attribute__((address_space(3)))

// CDNA5 async global->LDS copy path (ASYNCcnt-tracked), guarded so the file
// compiles on toolchains where the builtin is absent (falls back to
// load->VGPR->ds_store staging).
#if defined(__HIP_DEVICE_COMPILE__) && defined(__has_builtin)
#if __has_builtin(__builtin_amdgcn_global_load_async_to_lds_b128)
#define HAVE_ASYNC_LDS 1
#endif
#endif
#ifndef HAVE_ASYNC_LDS
#define HAVE_ASYNC_LDS 0
#endif

__device__ __forceinline__ void wait_async0() {
#if defined(__HIP_DEVICE_COMPILE__)
#if __has_builtin(__builtin_amdgcn_s_wait_asynccnt)
    __builtin_amdgcn_s_wait_asynccnt(0);
#else
    asm volatile("s_wait_asynccnt 0" ::: "memory");
#endif
#endif
}

// ---------------------------------------------------------------------------
// Kernel 1: RMSNorm -> bf16 x_normed + per-row rms (f32)
// one block (256 threads) per row; each thread owns 8 contiguous columns
// ---------------------------------------------------------------------------
__global__ __launch_bounds__(256)
void rmsnorm_kernel(const float* __restrict__ x,
                    const float* __restrict__ nw,
                    __bf16* __restrict__ xn,
                    float* __restrict__ rms)
{
    const int row = blockIdx.x;
    const int tid = threadIdx.x;
    const int c0  = tid * 8;

    const float4 a = *(const float4*)(x + (size_t)row * DD + c0);
    const float4 b = *(const float4*)(x + (size_t)row * DD + c0 + 4);
    float vals[8] = {a.x, a.y, a.z, a.w, b.x, b.y, b.z, b.w};

    float s = 0.f;
    #pragma unroll
    for (int i = 0; i < 8; ++i) s += vals[i] * vals[i];

    __shared__ float red[256];
    red[tid] = s;
    __syncthreads();
    for (int off = 128; off > 0; off >>= 1) {
        if (tid < off) red[tid] += red[tid + off];
        __syncthreads();
    }
    const float r = rsqrtf(red[0] * (1.0f / (float)DD) + EPSV);
    if (tid == 0) rms[row] = r;

    const float4 w0 = *(const float4*)(nw + c0);
    const float4 w1 = *(const float4*)(nw + c0 + 4);
    const float ws[8] = {w0.x, w0.y, w0.z, w0.w, w1.x, w1.y, w1.z, w1.w};

    bf16x8 o;
    #pragma unroll
    for (int i = 0; i < 8; ++i) o[i] = (__bf16)(vals[i] * r * ws[i]);
    *(bf16x8*)(xn + (size_t)row * DD + c0) = o;
}

// ---------------------------------------------------------------------------
// Kernel 2: W (f32, [3D, D] row-major) -> bf16, 8 elems/thread
// ---------------------------------------------------------------------------
__global__ __launch_bounds__(256)
void cvt_w_kernel(const float* __restrict__ W, __bf16* __restrict__ Wb)
{
    const size_t i = ((size_t)blockIdx.x * 256 + threadIdx.x) * 8;
    const float4 a = *(const float4*)(W + i);
    const float4 b = *(const float4*)(W + i + 4);
    const float vals[8] = {a.x, a.y, a.z, a.w, b.x, b.y, b.z, b.w};
    bf16x8 o;
    #pragma unroll
    for (int k = 0; k < 8; ++k) o[k] = (__bf16)vals[k];
    *(bf16x8*)(Wb + i) = o;
}

// ---------------------------------------------------------------------------
// Kernel 3: fused  ctrl = xn @ W^T (+b)  ->  sigmoid/softplus/INL epilogue
// Workgroup: 256 threads (8 waves), tile = 128 rows (M) x 64 cols (d).
// Each d-column's three projections (W rows d, D+d, 2D+d) are computed by the
// same workgroup so the elementwise tail fuses in-register.
// Double-buffered LDS; async global->LDS staging when available.
// ---------------------------------------------------------------------------
__global__ __launch_bounds__(256)
void gemm_fused_kernel(const __bf16* __restrict__ xn,
                       const __bf16* __restrict__ wb,
                       const float* __restrict__ x,
                       const float* __restrict__ v,
                       const float* __restrict__ shift,
                       const float* __restrict__ scale,
                       const float* __restrict__ nw,
                       const float* __restrict__ mu,
                       const float* __restrict__ bias,
                       const float* __restrict__ rms,
                       float* __restrict__ out)
{
    constexpr int KB  = 64;        // K per LDS stage
    constexpr int AST = KB + 8;    // padded row stride (halves): 144B, 16B-aligned
    constexpr int BST = KB + 8;
#if HAVE_ASYNC_LDS
    constexpr int NBUF = 2;
#else
    constexpr int NBUF = 1;
#endif

    __shared__ __bf16 sA[NBUF][128 * AST];
    __shared__ __bf16 sB[NBUF][3 * 64 * BST];

    const int tid  = threadIdx.x;
    const int lane = tid & 31;
    const int l15  = lane & 15;
    const int hi   = lane >> 4;
    const int wave = tid >> 5;
    const int wm   = wave & 3;    // 4 wave-rows  * 32 M = 128 M
    const int wd   = wave >> 2;   // 2 wave-cols  * 32 d = 64 d

    const int m0 = blockIdx.y * 128;
    const int d0 = blockIdx.x * 64;

    v8f acc[3][2][2];
    #pragma unroll
    for (int j = 0; j < 3; ++j)
        #pragma unroll
        for (int a = 0; a < 2; ++a)
            #pragma unroll
            for (int c = 0; c < 2; ++c)
                #pragma unroll
                for (int i = 0; i < 8; ++i) acc[j][a][c][i] = 0.f;

#if HAVE_ASYNC_LDS
    // Issue ASYNCcnt-tracked copies: 16B per lane per op, LDS write bypasses VGPRs.
    auto stage = [&](int k0, int buf) {
        for (int t = tid; t < 1024; t += 256) {          // A: 128 rows x 8 chunks
            const int row = t >> 3, ch = t & 7;
            const __bf16* g = &xn[(size_t)(m0 + row) * KK + k0 + ch * 8];
            __bf16* l = &sA[buf][row * AST + ch * 8];
            __builtin_amdgcn_global_load_async_to_lds_b128(
                (AS1 v4i*)g, (AS3 v4i*)l, 0, 0);
        }
        for (int t = tid; t < 1536; t += 256) {          // B: 3 x 64 rows x 8 chunks
            const int row = t >> 3, ch = t & 7;
            const int mat = row >> 6, r = row & 63;
            const __bf16* g = &wb[(size_t)(mat * DD + d0 + r) * KK + k0 + ch * 8];
            __bf16* l = &sB[buf][row * BST + ch * 8];
            __builtin_amdgcn_global_load_async_to_lds_b128(
                (AS1 v4i*)g, (AS3 v4i*)l, 0, 0);
        }
    };
#else
    auto stage = [&](int k0, int buf) {
        for (int t = tid; t < 1024; t += 256) {
            const int row = t >> 3, ch = t & 7;
            *(uint4*)&sA[buf][row * AST + ch * 8] =
                *(const uint4*)&xn[(size_t)(m0 + row) * KK + k0 + ch * 8];
        }
        for (int t = tid; t < 1536; t += 256) {
            const int row = t >> 3, ch = t & 7;
            const int mat = row >> 6, r = row & 63;
            *(uint4*)&sB[buf][row * BST + ch * 8] =
                *(const uint4*)&wb[(size_t)(mat * DD + d0 + r) * KK + k0 + ch * 8];
        }
    };
#endif

    auto compute = [&](int buf) {
        const __bf16* __restrict__ A = sA[buf];
        const __bf16* __restrict__ Bm = sB[buf];
        #pragma unroll
        for (int kk = 0; kk < KB; kk += 32) {
            // A frag: lane hi=0 holds K {0..7,16..23}; hi=1 holds {8..15,24..31}
            v16bf afrag[2];
            #pragma unroll
            for (int mt = 0; mt < 2; ++mt) {
                const __bf16* p = &A[(wm * 32 + mt * 16 + l15) * AST + kk];
                bf16x8 c0 = *(const bf16x8*)(p + hi * 8);
                bf16x8 c1 = *(const bf16x8*)(p + 16 + hi * 8);
                #pragma unroll
                for (int i = 0; i < 8; ++i) { afrag[mt][i] = c0[i]; afrag[mt][i + 8] = c1[i]; }
            }
            #pragma unroll
            for (int j = 0; j < 3; ++j) {
                // B frag: lane hi=0 holds K {0..7,16..23}; hi=1 holds {8..15,24..31}
                v16bf bfrag[2];
                #pragma unroll
                for (int nt = 0; nt < 2; ++nt) {
                    const __bf16* p = &Bm[(j * 64 + wd * 32 + nt * 16 + l15) * BST + kk];
                    bf16x8 c0 = *(const bf16x8*)(p + hi * 8);
                    bf16x8 c1 = *(const bf16x8*)(p + 16 + hi * 8);
                    #pragma unroll
                    for (int i = 0; i < 8; ++i) { bfrag[nt][i] = c0[i]; bfrag[nt][i + 8] = c1[i]; }
                }
                #pragma unroll
                for (int mt = 0; mt < 2; ++mt)
                    #pragma unroll
                    for (int nt = 0; nt < 2; ++nt)
                        acc[j][mt][nt] = __builtin_amdgcn_wmma_f32_16x16x32_bf16(
                            false, afrag[mt], false, bfrag[nt],
                            (short)0, acc[j][mt][nt], false, false);
            }
        }
    };

    constexpr int NST = KK / KB;   // 32 stages

#if HAVE_ASYNC_LDS
    stage(0, 0);
    wait_async0();
    __syncthreads();
    for (int i = 0; i < NST; ++i) {
        if (i + 1 < NST) stage((i + 1) * KB, (i + 1) & 1);  // prefetch overlaps WMMA
        compute(i & 1);
        wait_async0();          // my async copies for stage i+1 done
        __syncthreads();        // everyone done reading buf (i&1) + writing next
    }
#else
    for (int i = 0; i < NST; ++i) {
        __syncthreads();
        stage(i * KB, 0);
        __syncthreads();
        compute(0);
    }
#endif

    // ---------------- fused epilogue ----------------
    const int batch = m0 / SS;                 // 128 | 4096 -> constant per block
    const size_t TOT = (size_t)MM * DD;

    #pragma unroll
    for (int mt = 0; mt < 2; ++mt) {
        #pragma unroll
        for (int nt = 0; nt < 2; ++nt) {
            const int d = d0 + wd * 32 + nt * 16 + l15;
            const float sh  = shift[batch * DD + d];
            const float sc  = scale[batch * DD + d];
            const float muv = mu[d];
            const float nwv = nw[d];
            const float ba  = bias[d];
            const float bb_ = bias[DD + d];
            const float bg  = bias[2 * DD + d];
            const int mb = m0 + wm * 32 + mt * 16 + hi * 8;
            #pragma unroll
            for (int r = 0; r < 8; ++r) {
                const int m = mb + r;
                const size_t idx = (size_t)m * DD + d;
                const float ca = acc[0][mt][nt][r] + ba;
                const float cb = acc[1][mt][nt][r] + bb_;
                const float cg = acc[2][mt][nt][r] + bg;
                const float alpha = 1.f / (1.f + __expf(-ca));
                const float beta  = fmaxf(cb, 0.f) + log1pf(__expf(-fabsf(cb)));
                const float gate  = 1.f / (1.f + __expf(-cg));
                const float xv  = x[idx];
                const float xnv = xv * rms[m] * nwv;               // f32 re-norm
                const float xadaln = xnv * (1.f + sc) + sh;
                const float err = xadaln - muv;
                const float vnext = alpha * v[idx] - beta * err;
                out[idx]       = xv + xadaln + DTV * gate * vnext; // out
                out[TOT + idx] = vnext;                            // v_next
            }
        }
    }
}

// ---------------------------------------------------------------------------
// Launch
// ---------------------------------------------------------------------------
extern "C" void kernel_launch(void* const* d_in, const int* in_sizes, int n_in,
                              void* d_out, int out_size, void* d_ws, size_t ws_size,
                              hipStream_t stream)
{
    const float* x     = (const float*)d_in[0];
    const float* v     = (const float*)d_in[1];
    const float* shift = (const float*)d_in[2];
    const float* scale = (const float*)d_in[3];
    const float* nw    = (const float*)d_in[4];
    const float* mu    = (const float*)d_in[5];
    const float* W     = (const float*)d_in[6];
    const float* bias  = (const float*)d_in[7];
    float* out = (float*)d_out;

    char* ws = (char*)d_ws;
    __bf16* xn  = (__bf16*)ws;                                   // MM*KK*2 = 64 MiB
    __bf16* wbp = (__bf16*)(ws + (size_t)MM * KK * 2);           // N3*KK*2 = 24 MiB
    float*  rms = (float*)(ws + (size_t)MM * KK * 2 + (size_t)N3 * KK * 2);  // MM*4

    rmsnorm_kernel<<<MM, 256, 0, stream>>>(x, nw, xn, rms);
    cvt_w_kernel<<<(N3 * KK) / (256 * 8), 256, 0, stream>>>(W, wbp);

    dim3 grid(DD / 64, MM / 128);
    gemm_fused_kernel<<<grid, 256, 0, stream>>>(xn, wbp, x, v, shift, scale,
                                                nw, mu, bias, rms, out);
}
